// Attention_41549513621888
// MI455X (gfx1250) — matrix operand order repaired and checked by reading, not compiled, not run on hardware
//
#include <hip/hip_runtime.h>

// ---------------------------------------------------------------------------
// MI455X (gfx1250) sliding-window attention block, built on V_WMMA_F32_16X16X32_F16.
// Pipeline: [QKV projection (+fused RoPE, V transposed)] -> [flash attention] -> [out proj]
// Cross-lane softmax reductions use DPP16 (quad_perm/mirror) instead of ds_bpermute.
// ---------------------------------------------------------------------------

typedef __attribute__((ext_vector_type(16))) _Float16 v16h;
typedef __attribute__((ext_vector_type(8)))  _Float16 v8h;
typedef __attribute__((ext_vector_type(8)))  float    v8f;

#define DIMK   1024
#define SEQ    2048
#define NHEAD  16
#define HDIM   64
#define WIN    512
#define NROW   4096   // b * s

#define DPP_QUAD_XOR1   0xB1   // quad_perm [1,0,3,2]
#define DPP_QUAD_XOR2   0x4E   // quad_perm [2,3,0,1]
#define DPP_HALF_MIRROR 0x141  // row_half_mirror (swap 4<->4 within 8)
#define DPP_ROW_MIRROR  0x140  // row_mirror (swap 8<->8 within 16)

__device__ __forceinline__ v8f zero8() {
    v8f r;
#pragma unroll
    for (int i = 0; i < 8; ++i) r[i] = 0.0f;
    return r;
}

template <int CTRL>
__device__ __forceinline__ float dpp_movf(float x) {
    int i = __float_as_int(x);
    int r = __builtin_amdgcn_update_dpp(i, i, CTRL, 0xf, 0xf, true);
    return __int_as_float(r);
}

// max/sum reduction across each 16-lane half (never crosses the half boundary)
__device__ __forceinline__ float redmax16(float x) {
    x = fmaxf(x, dpp_movf<DPP_QUAD_XOR1>(x));
    x = fmaxf(x, dpp_movf<DPP_QUAD_XOR2>(x));
    x = fmaxf(x, dpp_movf<DPP_HALF_MIRROR>(x));
    x = fmaxf(x, dpp_movf<DPP_ROW_MIRROR>(x));
    return x;
}
__device__ __forceinline__ float redsum16(float x) {
    x += dpp_movf<DPP_QUAD_XOR1>(x);
    x += dpp_movf<DPP_QUAD_XOR2>(x);
    x += dpp_movf<DPP_HALF_MIRROR>(x);
    x += dpp_movf<DPP_ROW_MIRROR>(x);
    return x;
}

__device__ __forceinline__ v16h cat16(v8h lo, v8h hi) {
    return __builtin_shufflevector(lo, hi, 0,1,2,3,4,5,6,7,8,9,10,11,12,13,14,15);
}

// A fragment (16 rows x K=32, f16). p -> element (row0, k0); ld in halves.
// ISA layout: lane<16 holds row=lane, K {0..7,16..23}; lane>=16 row=lane-16, K {8..15,24..31}.
__device__ __forceinline__ v16h load_a16x32(const _Float16* p, int ld) {
    const int lane = threadIdx.x & 31;
    const _Float16* r = p + (lane & 15) * ld + ((lane >> 4) << 3);
    v8h lo = *(const v8h*)(r);
    v8h hi = *(const v8h*)(r + 16);
    return cat16(lo, hi);
}

// B fragment (K=32 x 16 cols, f16). Column n contiguous along K in memory:
// element (k, n) at p[n*ld + k]. lane<16: col=lane, K {0..7,16..23}; lane>=16: col=lane-16, K {8..15,24..31}.
__device__ __forceinline__ v16h load_b32x16(const _Float16* p, int ld) {
    const int lane = threadIdx.x & 31;
    const _Float16* r = p + (lane & 15) * ld + ((lane >> 4) << 3);
    v8h lo = *(const v8h*)(r);
    v8h hi = *(const v8h*)(r + 16);
    return cat16(lo, hi);
}

__device__ __forceinline__ v8f wmma16(v16h a, v16h b, v8f c) {
    return __builtin_amdgcn_wmma_f32_16x16x32_f16(false, a, false, b, (short)0, c, false, false);
}

// ---------------------------------------------------------------------------
// Projection GEMM: Out[m,n] = sum_k X[m,k] * W[n,k]   (X: NROWxDIMK, W: 1024xDIMK)
// One wave computes a 32(M) x 64(N) tile: 2 A-frags, 4 shared B-frags, 8 WMMA/k-step.
// DO_ROPE applies rotary to the f32 tile (DPP lane^1 partner exchange).
// TRANS_OUT stores V transposed as Vt[(b*1024+n)*SEQ + t] for contiguous key access.
// ---------------------------------------------------------------------------
template <int DO_ROPE, int TRANS_OUT>
__global__ void proj_rope_kernel(const _Float16* __restrict__ X,
                                 const _Float16* __restrict__ W,
                                 _Float16* __restrict__ Out) {
    const int lane  = threadIdx.x & 31;
    const int wid   = threadIdx.x >> 5;
    const int gw    = blockIdx.x * 4 + wid;     // 2048 waves total
    const int mBase = (gw & 127) * 32;          // 128 M tiles of 32
    const int nBase = (gw >> 7) * 64;           // 16 N super-tiles of 64
    const int hi    = lane >> 4;

    v8f acc[2][4];
#pragma unroll
    for (int mt = 0; mt < 2; ++mt)
#pragma unroll
        for (int t = 0; t < 4; ++t) acc[mt][t] = zero8();

    for (int k = 0; k < DIMK; k += 32) {
        const v16h a0 = load_a16x32(X + mBase * DIMK + k, DIMK);
        const v16h a1 = load_a16x32(X + (mBase + 16) * DIMK + k, DIMK);
#pragma unroll
        for (int t = 0; t < 4; ++t) {
            const v16h b = load_b32x16(W + (nBase + t * 16) * DIMK + k, DIMK);
            acc[0][t] = wmma16(a0, b, acc[0][t]);
            acc[1][t] = wmma16(a1, b, acc[1][t]);
        }
    }

    if (DO_ROPE) {
#pragma unroll
        for (int t = 0; t < 4; ++t) {
            const int n   = nBase + t * 16 + (lane & 15);
            const int dd  = n & (HDIM - 1);
            const float e = (float)(dd & ~1) * (1.0f / (float)HDIM);
            const float invf = __powf(10000.0f, -e);
            const float sgn  = (dd & 1) ? 1.0f : -1.0f;
#pragma unroll
            for (int mt = 0; mt < 2; ++mt)
#pragma unroll
            for (int v = 0; v < 8; ++v) {
                const int m   = mBase + mt * 16 + v + hi * 8;
                const float ang = (float)(m & (SEQ - 1)) * invf;
                float s, c;
                __sincosf(ang, &s, &c);
                const float x = acc[mt][t][v];
                const float partner = dpp_movf<DPP_QUAD_XOR1>(x);  // even<->odd column
                acc[mt][t][v] = x * c + sgn * partner * s;
            }
        }
    }

    if (TRANS_OUT) {
        const int b = mBase >> 11;                  // mBase / SEQ
#pragma unroll
        for (int mt = 0; mt < 2; ++mt) {
            const int t0 = ((mBase + mt * 16) & (SEQ - 1)) + hi * 8;
#pragma unroll
            for (int t = 0; t < 4; ++t) {
                const int n = nBase + t * 16 + (lane & 15);
                v8h pk;
#pragma unroll
                for (int v = 0; v < 8; ++v) pk[v] = (_Float16)acc[mt][t][v];
                *(v8h*)(Out + (b * DIMK + n) * SEQ + t0) = pk;   // one b128 store / frag
            }
        }
    } else {
#pragma unroll
        for (int mt = 0; mt < 2; ++mt)
#pragma unroll
        for (int t = 0; t < 4; ++t) {
            const int n = nBase + t * 16 + (lane & 15);
#pragma unroll
            for (int v = 0; v < 8; ++v)
                Out[(mBase + mt * 16 + v + hi * 8) * DIMK + n] = (_Float16)acc[mt][t][v];
        }
    }
}

// ---------------------------------------------------------------------------
// Flash attention with sliding window. One wave per (b, h, 16-query tile).
// S = Q K^T (2 WMMA per 16-key subtile), online softmax (DPP reductions),
// P staged through LDS into A-layout, O += P V (V pre-transposed).
// ---------------------------------------------------------------------------
__global__ void attn_window_kernel(const _Float16* __restrict__ Qh,
                                   const _Float16* __restrict__ Kh,
                                   const _Float16* __restrict__ Vt,
                                   _Float16* __restrict__ Oa) {
    __shared__ __align__(16) _Float16 pstage[4][16][32];   // 1KB per wave

    const int lane  = threadIdx.x & 31;
    const int wid   = threadIdx.x >> 5;
    const int gw    = blockIdx.x * 4 + wid;     // 4096 waves: 32 (b,h) x 128 qtiles
    const int qTile = gw & 127;
    const int bh    = gw >> 7;
    const int b     = bh >> 4;
    const int h     = bh & 15;
    const int hi    = lane >> 4;
    const int qLoc  = qTile * 16;               // first query row (in-sequence)
    const int qRow  = b * SEQ + qLoc;           // global row

    // Q fragments for d = 0..31 and 32..63 (loaded once)
    const _Float16* qp = Qh + qRow * DIMK + h * HDIM;
    const v16h qa0 = load_a16x32(qp, DIMK);
    const v16h qa1 = load_a16x32(qp + 32, DIMK);

    v8f o0 = zero8(), o1 = zero8(), o2 = zero8(), o3 = zero8();
    float mstate[8], lstate[8];
#pragma unroll
    for (int v = 0; v < 8; ++v) { mstate[v] = -1e30f; lstate[v] = 0.0f; }

    const int jStart = (qLoc >= (WIN - 1)) ? (((qLoc - (WIN - 1)) >> 5) << 5) : 0;
    const float scale = 0.125f;                 // 1/sqrt(HDIM)

    for (int jB = jStart; jB <= qLoc + 15; jB += 32) {
        // ---- S = Q K^T for keys [jB, jB+31] ----
        const _Float16* kp0 = Kh + (b * SEQ + jB) * DIMK + h * HDIM;
        const _Float16* kp1 = kp0 + 16 * DIMK;
        v8f s0 = zero8(), s1 = zero8();
        s0 = wmma16(qa0, load_b32x16(kp0, DIMK), s0);
        s0 = wmma16(qa1, load_b32x16(kp0 + 32, DIMK), s0);
        s1 = wmma16(qa0, load_b32x16(kp1, DIMK), s1);
        s1 = wmma16(qa1, load_b32x16(kp1 + 32, DIMK), s1);

        // ---- mask + online softmax (rows striped: row = v + hi*8) ----
        const int j0 = jB + (lane & 15);
        const int j1 = j0 + 16;
#pragma unroll
        for (int v = 0; v < 8; ++v) {
            const int i = qLoc + hi * 8 + v;
            const bool k0ok = (j0 <= i) && (j0 > i - WIN);
            const bool k1ok = (j1 <= i) && (j1 > i - WIN);
            const float a0 = k0ok ? s0[v] * scale : -1e30f;
            const float a1 = k1ok ? s1[v] * scale : -1e30f;

            const float t  = redmax16(fmaxf(a0, a1));      // DPP 16-lane reduction
            const float mn  = fmaxf(mstate[v], t);
            const float fac = __expf(mstate[v] - mn);
            const float e0  = k0ok ? __expf(a0 - mn) : 0.0f;
            const float e1  = k1ok ? __expf(a1 - mn) : 0.0f;

            const float r = redsum16(e0 + e1);             // DPP 16-lane reduction

            lstate[v] = lstate[v] * fac + r;
            mstate[v] = mn;
            o0[v] *= fac; o1[v] *= fac; o2[v] *= fac; o3[v] *= fac;

            pstage[wid][hi * 8 + v][lane & 15]        = (_Float16)e0;
            pstage[wid][hi * 8 + v][(lane & 15) + 16] = (_Float16)e1;
        }

        asm volatile("s_wait_dscnt 0" ::: "memory");   // LDS RAW fence (same-wave stage)

        // ---- O += P V  (P: 16x32 A-frag from LDS, V: key-contiguous transposed) ----
        const v16h pa = load_a16x32(&pstage[wid][0][0], 32);
        const _Float16* vp = Vt + (b * DIMK + h * HDIM) * SEQ + jB;
        o0 = wmma16(pa, load_b32x16(vp,            SEQ), o0);
        o1 = wmma16(pa, load_b32x16(vp + 16 * SEQ, SEQ), o1);
        o2 = wmma16(pa, load_b32x16(vp + 32 * SEQ, SEQ), o2);
        o3 = wmma16(pa, load_b32x16(vp + 48 * SEQ, SEQ), o3);
    }

    // ---- normalize and store (b, s, h*d) row-major ----
    _Float16* op = Oa + qRow * DIMK + h * HDIM;
#pragma unroll
    for (int v = 0; v < 8; ++v) {
        const float inv = 1.0f / lstate[v];
        const int row = v + hi * 8;
        const int col = lane & 15;
        op[row * DIMK + col]      = (_Float16)(o0[v] * inv);
        op[row * DIMK + col + 16] = (_Float16)(o1[v] * inv);
        op[row * DIMK + col + 32] = (_Float16)(o2[v] * inv);
        op[row * DIMK + col + 48] = (_Float16)(o3[v] * inv);
    }
}

// ---------------------------------------------------------------------------
extern "C" void kernel_launch(void* const* d_in, const int* in_sizes, int n_in,
                              void* d_out, int out_size, void* d_ws, size_t ws_size,
                              hipStream_t stream) {
    const _Float16* q  = (const _Float16*)d_in[0];
    const _Float16* k  = (const _Float16*)d_in[1];
    const _Float16* v  = (const _Float16*)d_in[2];
    const _Float16* wq = (const _Float16*)d_in[3];
    const _Float16* wk = (const _Float16*)d_in[4];
    const _Float16* wv = (const _Float16*)d_in[5];
    const _Float16* wo = (const _Float16*)d_in[6];
    _Float16* out = (_Float16*)d_out;

    _Float16* Qh = (_Float16*)d_ws;                 // (4096,1024) roped Q
    _Float16* Kh = Qh + (size_t)NROW * DIMK;        // (4096,1024) roped K
    _Float16* Vt = Kh + (size_t)NROW * DIMK;        // ((b*1024+n), 2048) transposed V
    _Float16* Oa = Vt + (size_t)NROW * DIMK;        // (4096,1024) attention output

    const dim3 blk(128);        // 4 wave32

    proj_rope_kernel<1, 0><<<dim3(512),  blk, 0, stream>>>(q, wq, Qh);
    proj_rope_kernel<1, 0><<<dim3(512),  blk, 0, stream>>>(k, wk, Kh);
    proj_rope_kernel<0, 1><<<dim3(512),  blk, 0, stream>>>(v, wv, Vt);
    attn_window_kernel    <<<dim3(1024), blk, 0, stream>>>(Qh, Kh, Vt, Oa);
    proj_rope_kernel<0, 0><<<dim3(512),  blk, 0, stream>>>(Oa, wo, out);
}
